// MultiHeadAttention_73143293051616
// MI455X (gfx1250) — matrix operand order repaired and checked by reading, not compiled, not run on hardware
//
#include <hip/hip_runtime.h>
#include <hip/hip_bf16.h>

// ---------------------------------------------------------------------------
// MI455X (gfx1250) implementation: LayerNorm -> QKV proj (WMMA bf16) ->
// L1-distance scores (VALU) -> softmax -> attn@V (WMMA) -> FC+residual (WMMA)
// ---------------------------------------------------------------------------

typedef __bf16 bf16;
typedef __attribute__((ext_vector_type(16))) __bf16 v16bf;
typedef __attribute__((ext_vector_type(8)))  __bf16 v8bf;
typedef __attribute__((ext_vector_type(8)))  float  v8f;

#define B_   4
#define L_   512
#define IN_  256
#define H_   8
#define DK_  32
#define DV_  32
#define INV_TEMP 0.17677669529663687f   // 1/sqrt(32)

__device__ __forceinline__ v8f wmma_bf16(v16bf a, v16bf b, v8f c) {
  return __builtin_amdgcn_wmma_f32_16x16x32_bf16(false, a, false, b, (short)0, c,
                                                 false, false);
}

// A-matrix fragment, 16x32 bf16 tile.  ISA layout: lane m = lane%16,
// half = lane/16; elems 0..7 = K[half*8 .. half*8+7], elems 8..15 =
// K[16+half*8 .. 16+half*8+7].  Two contiguous 16B loads per lane.
__device__ __forceinline__ v16bf load_a_frag(const bf16* tile, int ldk, int lane, int k0) {
  int m = lane & 15, half = lane >> 4;
  const bf16* row = tile + (size_t)m * ldk + k0 + half * 8;
  v8bf c0 = *(const v8bf*)(row);
  v8bf c1 = *(const v8bf*)(row + 16);
  v16bf a;
#pragma unroll
  for (int i = 0; i < 8; ++i) { a[i] = c0[i]; a[i + 8] = c1[i]; }
  return a;
}

// B-matrix fragment, 32x16 bf16 tile, sourced from TRANSPOSED storage
// baseT[n][k] (k contiguous).  Lane n = lane%16, khalf = lane/16; elems
// 0..7 = K[k0 + khalf*8 .. +7], elems 8..15 = K[k0 + 16 + khalf*8 .. +7].  Two 16B loads per lane.
__device__ __forceinline__ v16bf load_b_frag(const bf16* tileT, int ldk, int lane, int k0) {
  int n = lane & 15, khalf = lane >> 4;
  const bf16* row = tileT + (size_t)n * ldk + k0 + khalf * 8;
  v8bf c0 = *(const v8bf*)(row);
  v8bf c1 = *(const v8bf*)(row + 16);
  v16bf b;
#pragma unroll
  for (int i = 0; i < 8; ++i) { b[i] = c0[i]; b[i + 8] = c1[i]; }
  return b;
}

// ---------------------------------------------------------------------------
// K0: convert the four 256x256 f32 weights to bf16, transposed ([out][in]).
__global__ void convert_weights_kernel(const float* wq, const float* wk,
                                       const float* wv, const float* wfc,
                                       bf16* wqT, bf16* wkT, bf16* wvT, bf16* wfcT) {
  int idx = blockIdx.x * 256 + threadIdx.x;      // 0 .. 4*65536-1
  int which = idx >> 16;
  int e = idx & 65535;
  int kin = e >> 8, nout = e & 255;
  const float* src = which == 0 ? wq : which == 1 ? wk : which == 2 ? wv : wfc;
  bf16* dst = which == 0 ? wqT : which == 1 ? wkT : which == 2 ? wvT : wfcT;
  dst[(size_t)nout * 256 + kin] = (bf16)src[(size_t)kin * 256 + nout];
}

// ---------------------------------------------------------------------------
// K1: LayerNorm over last dim (256).  One block (256 threads) per row.
__global__ void layernorm_kernel(const float* __restrict__ x,
                                 const float* __restrict__ gamma,
                                 const float* __restrict__ beta,
                                 float* __restrict__ xn_f32,
                                 bf16* __restrict__ xn_bf16) {
  int row = blockIdx.x;
  int t = threadIdx.x;
  __shared__ float s1[256], s2[256];
  float v = x[(size_t)row * IN_ + t];
  s1[t] = v; s2[t] = v * v;
  __syncthreads();
  for (int off = 128; off > 0; off >>= 1) {
    if (t < off) { s1[t] += s1[t + off]; s2[t] += s2[t + off]; }
    __syncthreads();
  }
  float mean = s1[0] * (1.0f / IN_);
  float var  = s2[0] * (1.0f / IN_) - mean * mean;
  float rstd = rsqrtf(var + 1e-6f);
  float xn = (v - mean) * rstd * gamma[t] + beta[t];
  xn_f32[(size_t)row * IN_ + t] = xn;
  xn_bf16[(size_t)row * IN_ + t] = (bf16)xn;
}

__global__ void zero_kernel(float* p, int n) {
  int i = blockIdx.x * 256 + threadIdx.x;
  if (i < n) p[i] = 0.0f;
}

// ---------------------------------------------------------------------------
// K2: QKV projection GEMM.  M=2048 (b*L+l), N=768 (q|k|v), K=256, bf16 WMMA.
// grid (128, 12), block 128 (4 waves); each wave owns one 16x16 N-tile.
// q written f32 scaled by 1/sqrt(DK); k written f32; v written bf16
// TRANSPOSED to [b,h,d,l] so it can serve directly as the B operand later.
__global__ void proj_gemm_kernel(const bf16* __restrict__ xn,
                                 const bf16* __restrict__ wqT,
                                 const bf16* __restrict__ wkT,
                                 const bf16* __restrict__ wvT,
                                 float* __restrict__ qf,
                                 float* __restrict__ kf,
                                 bf16* __restrict__ vT) {
  int lane = threadIdx.x & 31;
  int wave = threadIdx.x >> 5;
  int m0 = blockIdx.x * 16;
  int ntile = blockIdx.y * 4 + wave;     // 0..47
  int n0g = ntile * 16;                  // global column in 0..767
  int which = n0g >> 8;                  // 0=q 1=k 2=v
  int n0 = n0g & 255;
  const bf16* wT = which == 0 ? wqT : which == 1 ? wkT : wvT;

  const bf16* abase = xn + (size_t)m0 * IN_;
  const bf16* bbase = wT + (size_t)n0 * IN_;
  v8f acc = {};
#pragma unroll
  for (int ks = 0; ks < 8; ++ks) {
    v16bf a = load_a_frag(abase, IN_, lane, ks * 32);
    v16bf b = load_b_frag(bbase, IN_, lane, ks * 32);
    acc = wmma_bf16(a, b, acc);
  }
  int nl = lane & 15;
  int mhi = (lane >> 4) * 8;
#pragma unroll
  for (int r = 0; r < 8; ++r) {
    int m = m0 + mhi + r;
    int ncol = n0 + nl;
    float val = acc[r];
    if (which == 0) {
      qf[(size_t)m * 256 + ncol] = val * INV_TEMP;
    } else if (which == 1) {
      kf[(size_t)m * 256 + ncol] = val;
    } else {
      int bb = m >> 9, l = m & 511;
      int hh = ncol >> 5, dd = ncol & 31;
      vT[(((size_t)(bb * H_ + hh)) * DV_ + dd) * L_ + l] = (bf16)val;
    }
  }
}

// ---------------------------------------------------------------------------
// K3: fused attention per (b, h, 16-row i-tile).  FOUR waves per block share
// one 50KB LDS tile set (scores + bf16 attn), quadrupling resident waves per
// WGP versus a 1-wave block — this is the VALU-bound stage, so occupancy is
// what hides the LDS-broadcast latency of the |q-k| loop.
//   scores(i,j) = mean_d |q(j,d) - k(i,d)|   (q pre-scaled by 1/sqrt(DK))
//   softmax over j; column sums atomically accumulated for the attention map;
//   out(i,d) = sum_j attn(i,j) * v(j,d) via bf16 WMMA (waves 0/1, one N-tile each).
__global__ void attn_kernel(const float* __restrict__ qf,
                            const float* __restrict__ kf,
                            const bf16* __restrict__ vT,
                            bf16* __restrict__ attn_out,
                            float* __restrict__ am_accum) {
  int itile = blockIdx.x, h = blockIdx.y, b = blockIdx.z;
  int tid  = threadIdx.x;
  int lane = tid & 31;
  int wave = tid >> 5;                    // 0..3
  int i0 = itile * 16;

  __shared__ __align__(16) float k_lds[16][DK_];      //  2 KB
  __shared__ __align__(16) float sc[16][L_];          // 32 KB
  __shared__ __align__(16) bf16  at[16][L_];          // 16 KB
  __shared__ float rmax[16], rinvsum[16];

  // stage the 16 k-rows for this i-tile (invariant over j)
  for (int t = tid; t < 16 * DK_; t += 128) {
    int i = t >> 5, d = t & 31;
    k_lds[i][d] = kf[((size_t)(b * L_ + i0 + i)) * 256 + h * DK_ + d];
  }
  __syncthreads();

  // scores: wave w owns j-range [w*128, w*128+128); each lane one j per chunk
  int jbeg = wave * 128, jend = jbeg + 128;
  for (int jc = jbeg; jc < jend; jc += 32) {
    int j = jc + lane;
    const float* qrow = qf + ((size_t)(b * L_ + j)) * 256 + h * DK_;
    if (jc + 32 < jend)                       // hide next q-row fetch
      __builtin_prefetch(qrow + 32 * 256, 0, 0);  // -> global_prefetch_b8
    float qv[DK_];
#pragma unroll
    for (int d = 0; d < DK_; ++d) qv[d] = qrow[d];
#pragma unroll
    for (int i = 0; i < 16; ++i) {
      float s = 0.0f;
#pragma unroll
      for (int d = 0; d < DK_; ++d) s += fabsf(qv[d] - k_lds[i][d]);
      sc[i][j] = s * (1.0f / DK_);
    }
  }
  __syncthreads();

  // per-row softmax statistics: wave w owns rows [w*4, w*4+4)
  for (int i = wave * 4; i < wave * 4 + 4; ++i) {
    float m = -1e30f;
    for (int j = lane; j < L_; j += 32) m = fmaxf(m, sc[i][j]);
#pragma unroll
    for (int off = 16; off > 0; off >>= 1) m = fmaxf(m, __shfl_xor(m, off, 32));
    float s = 0.0f;
    for (int j = lane; j < L_; j += 32) s += __expf(sc[i][j] - m);
#pragma unroll
    for (int off = 16; off > 0; off >>= 1) s += __shfl_xor(s, off, 32);
    if (lane == 0) { rmax[i] = m; rinvsum[i] = 1.0f / s; }
  }
  __syncthreads();

  // normalized attention -> bf16 tile; column sums -> attention-map accum
  for (int jc = jbeg; jc < jend; jc += 32) {
    int j = jc + lane;
    float cs = 0.0f;
#pragma unroll
    for (int i = 0; i < 16; ++i) {
      float e = __expf(sc[i][j] - rmax[i]) * rinvsum[i];
      at[i][j] = (bf16)e;
      cs += e;
    }
    atomicAdd(&am_accum[b * L_ + j], cs);
  }
  __syncthreads();

  // attn(16 x 512) @ V(512 x 32): waves 0/1 take one 16x16 N-tile each
  if (wave < 2) {
    int nt = wave;
    const bf16* vbase = vT + ((size_t)(b * H_ + h)) * DV_ * L_ + (size_t)(nt * 16) * L_;
    v8f acc = {};
    for (int ks = 0; ks < 16; ++ks) {
      v16bf a  = load_a_frag(&at[0][0], L_, lane, ks * 32);
      v16bf bb = load_b_frag(vbase, L_, lane, ks * 32);
      acc = wmma_bf16(a, bb, acc);
    }
    int ncol = h * DV_ + nt * 16 + (lane & 15);
    int mrow = i0 + ((lane >> 4) * 8);
#pragma unroll
    for (int r = 0; r < 8; ++r) {
      attn_out[((size_t)(b * L_ + mrow + r)) * 256 + ncol] = (bf16)acc[r];
    }
  }
}

// ---------------------------------------------------------------------------
// K4: FC GEMM + residual.  M=2048, N=256, K=256, bf16 WMMA, f32 out.
__global__ void fc_gemm_kernel(const bf16* __restrict__ ao,
                               const bf16* __restrict__ wfcT,
                               const float* __restrict__ xn_f32,
                               float* __restrict__ out) {
  int lane = threadIdx.x & 31;
  int wave = threadIdx.x >> 5;
  int m0 = blockIdx.x * 16;
  int n0 = (blockIdx.y * 4 + wave) * 16;

  const bf16* abase = ao + (size_t)m0 * 256;
  const bf16* bbase = wfcT + (size_t)n0 * 256;
  v8f acc = {};
#pragma unroll
  for (int ks = 0; ks < 8; ++ks) {
    v16bf a = load_a_frag(abase, 256, lane, ks * 32);
    v16bf b = load_b_frag(bbase, 256, lane, ks * 32);
    acc = wmma_bf16(a, b, acc);
  }
  int nl = lane & 15;
  int mhi = (lane >> 4) * 8;
#pragma unroll
  for (int r = 0; r < 8; ++r) {
    size_t idx = (size_t)(m0 + mhi + r) * 256 + n0 + nl;
    out[idx] = acc[r] + xn_f32[idx];
  }
}

// ---------------------------------------------------------------------------
// K5: attention-map normalization.  am = (colsum - min) / (max - min).
// (mean over heads cancels in the affine normalization.)
__global__ void am_finalize_kernel(const float* __restrict__ am_accum,
                                   float* __restrict__ am_out) {
  int b = blockIdx.x, t = threadIdx.x;          // 256 threads, 2 j each
  __shared__ float smn[256], smx[256];
  float v0 = am_accum[b * L_ + t];
  float v1 = am_accum[b * L_ + 256 + t];
  smn[t] = fminf(v0, v1); smx[t] = fmaxf(v0, v1);
  __syncthreads();
  for (int off = 128; off > 0; off >>= 1) {
    if (t < off) {
      smn[t] = fminf(smn[t], smn[t + off]);
      smx[t] = fmaxf(smx[t], smx[t + off]);
    }
    __syncthreads();
  }
  float mn = smn[0], mx = smx[0];
  float inv = 1.0f / (mx - mn);
  am_out[b * L_ + t] = (v0 - mn) * inv;
  am_out[b * L_ + 256 + t] = (v1 - mn) * inv;
}

// ---------------------------------------------------------------------------
extern "C" void kernel_launch(void* const* d_in, const int* in_sizes, int n_in,
                              void* d_out, int out_size, void* d_ws, size_t ws_size,
                              hipStream_t stream) {
  const float* x     = (const float*)d_in[0];
  const float* w_q   = (const float*)d_in[1];
  const float* w_k   = (const float*)d_in[2];
  const float* w_v   = (const float*)d_in[3];
  const float* w_fc  = (const float*)d_in[4];
  const float* gamma = (const float*)d_in[5];
  const float* beta  = (const float*)d_in[6];

  char* ws = (char*)d_ws;
  size_t off = 0;
  auto alloc = [&](size_t bytes) -> void* {
    void* p = ws + off;
    off = (off + bytes + 255) & ~(size_t)255;
    return p;
  };
  float* xn_f32  = (float*)alloc((size_t)2048 * 256 * 4);
  bf16*  xn_bf16 = (bf16*) alloc((size_t)2048 * 256 * 2);
  bf16*  wqT     = (bf16*) alloc((size_t)256 * 256 * 2);
  bf16*  wkT     = (bf16*) alloc((size_t)256 * 256 * 2);
  bf16*  wvT     = (bf16*) alloc((size_t)256 * 256 * 2);
  bf16*  wfcT    = (bf16*) alloc((size_t)256 * 256 * 2);
  float* qf      = (float*)alloc((size_t)2048 * 256 * 4);
  float* kf      = (float*)alloc((size_t)2048 * 256 * 4);
  bf16*  vT      = (bf16*) alloc((size_t)B_ * H_ * DV_ * L_ * 2);
  bf16*  attn_o  = (bf16*) alloc((size_t)2048 * 256 * 2);
  float* am_acc  = (float*)alloc((size_t)B_ * L_ * 4);

  float* out = (float*)d_out;                 // [B, L, IN]  = 524288 floats
  float* am  = out + (size_t)B_ * L_ * IN_;   // [B, L]      =   2048 floats

  convert_weights_kernel<<<1024, 256, 0, stream>>>(w_q, w_k, w_v, w_fc,
                                                   wqT, wkT, wvT, wfcT);
  layernorm_kernel<<<2048, 256, 0, stream>>>(x, gamma, beta, xn_f32, xn_bf16);
  zero_kernel<<<(B_ * L_ + 255) / 256, 256, 0, stream>>>(am_acc, B_ * L_);
  proj_gemm_kernel<<<dim3(128, 12), 128, 0, stream>>>(xn_bf16, wqT, wkT, wvT,
                                                      qf, kf, vT);
  attn_kernel<<<dim3(L_ / 16, H_, B_), 128, 0, stream>>>(qf, kf, vT, attn_o, am_acc);
  fc_gemm_kernel<<<dim3(128, 4), 128, 0, stream>>>(attn_o, wfcT, xn_f32, out);
  am_finalize_kernel<<<B_, 256, 0, stream>>>(am_acc, am);
}
